// PCTransformer_68710886802223
// MI455X (gfx1250) — matrix operand order repaired and checked by reading, not compiled, not run on hardware
//
#include <hip/hip_runtime.h>
#include <hip/hip_bf16.h>
#include <math.h>

typedef __attribute__((ext_vector_type(16))) _Float16 v16h;
typedef __attribute__((ext_vector_type(8)))  _Float16 v8h;
typedef __attribute__((ext_vector_type(8)))  float    v8f;

#define V_    32000
#define D_    512
#define H_    8
#define L_    2
#define B_    4
#define S_    128
#define DFF_  2048
#define T_    3
#define HD_   64
#define M_    (B_*S_)      // 512 rows in every GEMM
#define GAMMA_ 0.1f
#define SCALE_ 0.125f      // 1/sqrt(HD)

// ---------------------------------------------------------------------------
// Staging helpers: 128 threads move a 64x32 f32 tile (512 float4s, 4 per thread)
// ---------------------------------------------------------------------------
__device__ __forceinline__ void ld_tile_rows(const float* __restrict__ P, int ldp,
                                             int rowBase, int k0, int tid, float4* f) {
    #pragma unroll
    for (int i = 0; i < 4; ++i) {
        int idx = i * 128 + tid;                 // 0..511
        int r   = idx >> 3;
        int c4  = (idx & 7) << 2;
        f[i] = *reinterpret_cast<const float4*>(&P[(size_t)(rowBase + r) * ldp + k0 + c4]);
    }
}
__device__ __forceinline__ void ld_tile_cols(const float* __restrict__ P, int ldp,
                                             int colBase, int k0, int tid, float4* f) {
    // P is [K,N]; read rows k, 4 consecutive n per float4
    #pragma unroll
    for (int i = 0; i < 4; ++i) {
        int idx = i * 128 + tid;
        int kk  = idx >> 4;
        int nn4 = (idx & 15) << 2;
        f[i] = *reinterpret_cast<const float4*>(&P[(size_t)(k0 + kk) * ldp + colBase + nn4]);
    }
}
__device__ __forceinline__ void st_tile_rows(_Float16 (*dst)[40], int tid, const float4* f) {
    #pragma unroll
    for (int i = 0; i < 4; ++i) {
        int idx = i * 128 + tid;
        int r   = idx >> 3;
        int c4  = (idx & 7) << 2;
        dst[r][c4 + 0] = (_Float16)f[i].x;  dst[r][c4 + 1] = (_Float16)f[i].y;
        dst[r][c4 + 2] = (_Float16)f[i].z;  dst[r][c4 + 3] = (_Float16)f[i].w;
    }
}
__device__ __forceinline__ void st_tile_trans(_Float16 (*dst)[40], int tid, const float4* f) {
    #pragma unroll
    for (int i = 0; i < 4; ++i) {
        int idx = i * 128 + tid;
        int kk  = idx >> 4;
        int nn4 = (idx & 15) << 2;
        dst[nn4 + 0][kk] = (_Float16)f[i].x;  dst[nn4 + 1][kk] = (_Float16)f[i].y;
        dst[nn4 + 2][kk] = (_Float16)f[i].z;  dst[nn4 + 3][kk] = (_Float16)f[i].w;
    }
}

// ---------------------------------------------------------------------------
// WMMA GEMM: C[M,N] = beta*C + alpha*(A[M,K] @ op(B)) + bias[N]
//   transB=1: B is [N,K] row-major (compute A @ B^T)
//   transB=0: B is [K,N] row-major (compute A @ B)
// Block: 128 threads = 4 waves. Tile 64x64, BK=32, double-buffered LDS
// (one barrier per k-step; global loads for tile kt+1 issued before the
// WMMAs of tile kt). Requires M%64==0, N%64==0, K%32==0 (true here).
// f32 -> f16 in LDS staging; f32 accumulation. Both LDS tiles stored
// K-contiguous so each fragment is exactly two ds_load_b128 per lane.
// ---------------------------------------------------------------------------
__global__ __launch_bounds__(128)
void gemm_wmma_kernel(const float* __restrict__ A, const float* __restrict__ B,
                      const float* __restrict__ bias, float* __restrict__ C,
                      int M, int N, int K, int transB, float alpha, float beta) {
    __shared__ _Float16 As[2][64][40];   // [buf][m][k]  (row stride 80B, 16B aligned)
    __shared__ _Float16 Bs[2][64][40];   // [buf][n][k]

    const int tid   = threadIdx.x;
    const int wave  = tid >> 5;
    const int lane  = tid & 31;
    const int lan15 = lane & 15;
    const int khalf = (lane >> 4);          // 0: lanes 0-15, 1: lanes 16-31
    const int mTile = blockIdx.y * 64;
    const int nTile = blockIdx.x * 64;
    const int mrow  = wave * 16 + lan15;

    v8f acc[4];
    #pragma unroll
    for (int s = 0; s < 4; ++s) acc[s] = (v8f){0,0,0,0,0,0,0,0};

    const int nk = K >> 5;
    float4 fa[4], fb[4];

    // ---- prologue: stage tile 0 into buffer 0 ----
    ld_tile_rows(A, K, mTile, 0, tid, fa);
    if (transB) ld_tile_rows(B, K, nTile, 0, tid, fb);
    else        ld_tile_cols(B, N, nTile, 0, tid, fb);
    st_tile_rows(As[0], tid, fa);
    if (transB) st_tile_rows(Bs[0], tid, fb);
    else        st_tile_trans(Bs[0], tid, fb);
    __syncthreads();

    int cur = 0;
    for (int kt = 0; kt < nk; ++kt) {
        const bool hasNext = (kt + 1) < nk;
        // ---- issue global loads for next tile (overlap with WMMAs) ----
        if (hasNext) {
            const int k0n = (kt + 1) << 5;
            ld_tile_rows(A, K, mTile, k0n, tid, fa);
            if (transB) ld_tile_rows(B, K, nTile, k0n, tid, fb);
            else        ld_tile_cols(B, N, nTile, k0n, tid, fb);
            if (kt + 2 < nk) {  // L2 hint for tile kt+2 (global_prefetch_b8)
                __builtin_prefetch(&A[(size_t)(mTile + (tid >> 1)) * K + ((kt + 2) << 5)], 0, 1);
            }
        }

        // ---- fragments from current buffer ----
        const _Float16* arow = &As[cur][mrow][0];
        const v8h a_lo = *reinterpret_cast<const v8h*>(arow + 16 * khalf);
        const v8h a_hi = *reinterpret_cast<const v8h*>(arow + 16 * khalf + 8);
        const v16h af  = __builtin_shufflevector(a_lo, a_hi,
                            0,1,2,3,4,5,6,7,8,9,10,11,12,13,14,15);
        v16h bf4[4];
        #pragma unroll
        for (int s = 0; s < 4; ++s) {
            const _Float16* brow = &Bs[cur][s * 16 + lan15][0];
            const v8h b_lo = *reinterpret_cast<const v8h*>(brow + 16 * khalf);
            const v8h b_hi = *reinterpret_cast<const v8h*>(brow + 16 * khalf + 8);
            bf4[s] = __builtin_shufflevector(b_lo, b_hi,
                        0,1,2,3,4,5,6,7,8,9,10,11,12,13,14,15);
        }
        #pragma unroll
        for (int s = 0; s < 4; ++s) {
            acc[s] = __builtin_amdgcn_wmma_f32_16x16x32_f16(
                false, af, false, bf4[s], (short)0, acc[s], false, false);
        }

        // ---- stage next tile into the other buffer ----
        if (hasNext) {
            st_tile_rows(As[cur ^ 1], tid, fa);
            if (transB) st_tile_rows(Bs[cur ^ 1], tid, fb);
            else        st_tile_trans(Bs[cur ^ 1], tid, fb);
        }
        __syncthreads();
        cur ^= 1;
    }

    // ---- epilogue: C VGPR r -> row (r + 8*khalf), col = lan15 ----
    const int rowBase = mTile + wave * 16 + 8 * khalf;
    #pragma unroll
    for (int s = 0; s < 4; ++s) {
        const int col = nTile + s * 16 + lan15;
        const float bv = bias ? bias[col] : 0.0f;
        #pragma unroll
        for (int r = 0; r < 8; ++r) {
            size_t cidx = (size_t)(rowBase + r) * N + col;
            float v = alpha * acc[s][r] + bv;
            if (beta != 0.0f) v += beta * C[cidx];
            C[cidx] = v;
        }
    }
}

// ---------------------------------------------------------------------------
// Elementwise / reduction helpers
// ---------------------------------------------------------------------------
__global__ void layernorm_kernel(const float* __restrict__ x,
                                 const float* __restrict__ g,
                                 const float* __restrict__ b,
                                 float* __restrict__ out) {
    __shared__ float r1[128], r2[128];
    const int row = blockIdx.x, tid = threadIdx.x;
    float s = 0.f, s2 = 0.f;
    for (int c = tid; c < D_; c += 128) {
        float v = x[(size_t)row * D_ + c];
        s += v; s2 += v * v;
    }
    r1[tid] = s; r2[tid] = s2; __syncthreads();
    for (int o = 64; o > 0; o >>= 1) {
        if (tid < o) { r1[tid] += r1[tid + o]; r2[tid] += r2[tid + o]; }
        __syncthreads();
    }
    const float mu  = r1[0] / D_;
    const float var = r2[0] / D_ - mu * mu;
    const float rstd = rsqrtf(var + 1e-5f);
    for (int c = tid; c < D_; c += 128) {
        float v = x[(size_t)row * D_ + c];
        out[(size_t)row * D_ + c] = (v - mu) * rstd * g[c] + b[c];
    }
}

// in-place: logits -> onehot(target) - logits
__global__ void err_kernel(float* __restrict__ logits,
                           const int* __restrict__ target) {
    int idx = blockIdx.x * 256 + threadIdx.x;
    if (idx >= M_ * V_) return;
    int m = idx / V_, c = idx % V_;
    int t = target[m]; t = t < 0 ? 0 : (t >= V_ ? V_ - 1 : t);
    logits[idx] = ((c == t) ? 1.0f : 0.0f) - logits[idx];
}

__global__ void sub_kernel(const float* __restrict__ a, const float* __restrict__ b,
                           float* __restrict__ o, int n) {
    int idx = blockIdx.x * 256 + threadIdx.x;
    if (idx < n) o[idx] = a[idx] - b[idx];
}

// in-place on z: z -> (fc2 - gelu(z)) * gelu'(z)   (exact GELU)
__global__ void gelu_vjp_kernel(float* __restrict__ z1, const float* __restrict__ fc2,
                                int n) {
    int idx = blockIdx.x * 256 + threadIdx.x;
    if (idx >= n) return;
    float z = z1[idx];
    float cphi = 0.5f * (1.0f + erff(z * 0.70710678118f));
    float pdf  = 0.39894228040f * __expf(-0.5f * z * z);
    z1[idx] = (fc2[idx] - z * cphi) * (cphi + z * pdf);
}

// ---------------------------------------------------------------------------
// Causal attention forward + exact VJP (scalar; tiny problem)
// ---------------------------------------------------------------------------
__global__ void attn_softmax_kernel(const float* __restrict__ q,
                                    const float* __restrict__ k,
                                    float* __restrict__ p) {
    __shared__ float sh[128];
    const int blk = blockIdx.x;
    const int qr = blk % S_, h = (blk / S_) % H_, b = blk / (S_ * H_);
    const int j = threadIdx.x;
    float sc = -1e30f;
    if (j <= qr) {
        const float* qp = q + ((size_t)(b * S_ + qr)) * D_ + h * HD_;
        const float* kp = k + ((size_t)(b * S_ + j)) * D_ + h * HD_;
        float dot = 0.f;
        #pragma unroll 8
        for (int d = 0; d < HD_; ++d) dot += qp[d] * kp[d];
        sc = dot * SCALE_;
    }
    sh[j] = sc; __syncthreads();
    for (int o = 64; o > 0; o >>= 1) {
        if (j < o) sh[j] = fmaxf(sh[j], sh[j + o]);
        __syncthreads();
    }
    const float mx = sh[0]; __syncthreads();
    float e = (j <= qr) ? __expf(sc - mx) : 0.f;
    sh[j] = e; __syncthreads();
    for (int o = 64; o > 0; o >>= 1) {
        if (j < o) sh[j] += sh[j + o];
        __syncthreads();
    }
    p[(size_t)blk * S_ + j] = e / sh[0];
}

__global__ void attn_pv_kernel(const float* __restrict__ p, const float* __restrict__ v,
                               float* __restrict__ out) {
    int idx = blockIdx.x * 256 + threadIdx.x;
    if (idx >= M_ * D_) return;
    int c = idx % D_, srow = idx / D_;
    int qp = srow % S_, b = srow / S_, h = c / HD_;
    const float* prow = p + ((size_t)((b * H_ + h) * S_ + qp)) * S_;
    float acc = 0.f;
    for (int kk = 0; kk <= qp; ++kk)
        acc += prow[kk] * v[(size_t)(b * S_ + kk) * D_ + c];
    out[idx] = acc;
}

__global__ void attn_dv_kernel(const float* __restrict__ p, const float* __restrict__ g,
                               float* __restrict__ dv) {
    int idx = blockIdx.x * 256 + threadIdx.x;
    if (idx >= M_ * D_) return;
    int c = idx % D_, srow = idx / D_;
    int kp = srow % S_, b = srow / S_, h = c / HD_;
    float acc = 0.f;
    for (int qp = kp; qp < S_; ++qp)
        acc += p[((size_t)((b * H_ + h) * S_ + qp)) * S_ + kp] *
               g[(size_t)(b * S_ + qp) * D_ + c];
    dv[idx] = acc;
}

__global__ void attn_dp_kernel(const float* __restrict__ g, const float* __restrict__ v,
                               float* __restrict__ dp) {
    int idx = blockIdx.x * 256 + threadIdx.x;
    if (idx >= B_ * H_ * S_ * S_) return;
    int kp = idx % S_;  int t = idx / S_;
    int qp = t % S_;    t /= S_;
    int h  = t % H_;    int b = t / H_;
    float acc = 0.f;
    if (kp <= qp) {
        const float* gp = g + (size_t)(b * S_ + qp) * D_ + h * HD_;
        const float* vp = v + (size_t)(b * S_ + kp) * D_ + h * HD_;
        #pragma unroll 8
        for (int d = 0; d < HD_; ++d) acc += gp[d] * vp[d];
    }
    dp[idx] = acc;
}

// in-place: dp -> p * (dp - sum_k dp*p) * SCALE
__global__ void attn_dsoftmax_kernel(const float* __restrict__ p, float* __restrict__ dp) {
    __shared__ float sh[128];
    const int blk = blockIdx.x, j = threadIdx.x;
    const size_t base = (size_t)blk * S_;
    const float pj = p[base + j], dj = dp[base + j];
    sh[j] = pj * dj; __syncthreads();
    for (int o = 64; o > 0; o >>= 1) {
        if (j < o) sh[j] += sh[j + o];
        __syncthreads();
    }
    dp[base + j] = pj * (dj - sh[0]) * SCALE_;
}

__global__ void attn_dq_kernel(const float* __restrict__ ds, const float* __restrict__ k,
                               float* __restrict__ dq) {
    int idx = blockIdx.x * 256 + threadIdx.x;
    if (idx >= M_ * D_) return;
    int c = idx % D_, srow = idx / D_;
    int qp = srow % S_, b = srow / S_, h = c / HD_;
    const float* drow = ds + ((size_t)((b * H_ + h) * S_ + qp)) * S_;
    float acc = 0.f;
    for (int kk = 0; kk <= qp; ++kk)
        acc += drow[kk] * k[(size_t)(b * S_ + kk) * D_ + c];
    dq[idx] = acc;
}

__global__ void attn_dk_kernel(const float* __restrict__ ds, const float* __restrict__ q,
                               float* __restrict__ dk) {
    int idx = blockIdx.x * 256 + threadIdx.x;
    if (idx >= M_ * D_) return;
    int c = idx % D_, srow = idx / D_;
    int kp = srow % S_, b = srow / S_, h = c / HD_;
    float acc = 0.f;
    for (int qp = kp; qp < S_; ++qp)
        acc += ds[((size_t)((b * H_ + h) * S_ + qp)) * S_ + kp] *
               q[(size_t)(b * S_ + qp) * D_ + c];
    dk[idx] = acc;
}

// ---------------------------------------------------------------------------
// Host orchestration
// ---------------------------------------------------------------------------
extern "C" void kernel_launch(void* const* d_in, const int* in_sizes, int n_in,
                              void* d_out, int out_size, void* d_ws, size_t ws_size,
                              hipStream_t stream) {
    (void)in_sizes; (void)n_in; (void)out_size; (void)ws_size;
    const int*   target_ids = (const int*)d_in[1];
    const float* Wq   = (const float*)d_in[4];
    const float* bq   = (const float*)d_in[5];
    const float* Wk   = (const float*)d_in[6];
    const float* bk   = (const float*)d_in[7];
    const float* Wv   = (const float*)d_in[8];
    const float* bv   = (const float*)d_in[9];
    const float* Wo   = (const float*)d_in[10];
    const float* bo   = (const float*)d_in[11];
    const float* ln1g = (const float*)d_in[12];
    const float* ln1b = (const float*)d_in[13];
    const float* ln2g = (const float*)d_in[14];
    const float* ln2b = (const float*)d_in[15];
    const float* W1   = (const float*)d_in[16];
    const float* b1   = (const float*)d_in[17];
    const float* W2   = (const float*)d_in[18];
    const float* b2   = (const float*)d_in[19];
    const float* Wout = (const float*)d_in[20];
    const float* bout = (const float*)d_in[21];

    // workspace carve-up (all offsets are multiples of 4 floats -> 16B aligned)
    float* ws = (float*)d_ws;
    size_t off = 0;
    float* out_x  = ws + off; off += (size_t)M_ * D_;
    float* attn_x = ws + off; off += (size_t)L_ * M_ * D_;
    float* ao_x   = ws + off; off += (size_t)L_ * M_ * D_;
    float* fc1_x  = ws + off; off += (size_t)L_ * M_ * D_;
    float* fc2_x  = ws + off; off += (size_t)L_ * M_ * DFF_;
    const size_t zero_bytes = off * sizeof(float);
    float* tmpD  = ws + off; off += (size_t)M_ * D_;
    float* tmpD2 = ws + off; off += (size_t)M_ * D_;
    float* tmpF  = ws + off; off += (size_t)M_ * DFF_;
    float* errb  = ws + off; off += (size_t)M_ * V_;
    float* qb    = ws + off; off += (size_t)M_ * D_;
    float* kb    = ws + off; off += (size_t)M_ * D_;
    float* vb    = ws + off; off += (size_t)M_ * D_;
    float* dqb   = ws + off; off += (size_t)M_ * D_;
    float* dkb   = ws + off; off += (size_t)M_ * D_;
    float* dvb   = ws + off; off += (size_t)M_ * D_;
    float* pbuf  = ws + off; off += (size_t)B_ * H_ * S_ * S_;
    float* dsb   = ws + off; off += (size_t)B_ * H_ * S_ * S_;

    hipMemsetAsync(ws, 0, zero_bytes, stream);  // zero PC states each launch

    auto gemm = [&](const float* A, const float* Bm, const float* bias, float* C,
                    int Mm, int Nn, int Kk, bool transB, float alpha, float beta) {
        dim3 g(Nn / 64, Mm / 64);
        gemm_wmma_kernel<<<g, 128, 0, stream>>>(A, Bm, bias, C, Mm, Nn, Kk,
                                                transB ? 1 : 0, alpha, beta);
    };
    const int gMD  = (M_ * D_) / 256;
    const int gMF  = (M_ * DFF_) / 256;
    const int gP   = (B_ * H_ * S_ * S_) / 256;
    const int gErr = (M_ * V_ + 255) / 256;

    for (int t = 0; t < T_; ++t) {
        // output PC layer
        gemm(out_x, Wout, bout, errb, M_, V_, D_, true, 1.f, 0.f);       // logits
        err_kernel<<<gErr, 256, 0, stream>>>(errb, target_ids);          // onehot - logits
        gemm(errb, Wout, nullptr, out_x, M_, D_, V_, false, GAMMA_, 1.f);
        const float* nt = out_x;

        for (int i = L_ - 1; i >= 0; --i) {
            const float* Wq_i = Wq + (size_t)i * D_ * D_;   const float* bq_i = bq + i * D_;
            const float* Wk_i = Wk + (size_t)i * D_ * D_;   const float* bk_i = bk + i * D_;
            const float* Wv_i = Wv + (size_t)i * D_ * D_;   const float* bv_i = bv + i * D_;
            const float* Wo_i = Wo + (size_t)i * D_ * D_;   const float* bo_i = bo + i * D_;
            const float* W1_i = W1 + (size_t)i * DFF_ * D_; const float* b1_i = b1 + i * DFF_;
            const float* W2_i = W2 + (size_t)i * D_ * DFF_; const float* b2_i = b2 + i * D_;
            float* attn_x_i = attn_x + (size_t)i * M_ * D_;
            float* ao_x_i   = ao_x   + (size_t)i * M_ * D_;
            float* fc1_x_i  = fc1_x  + (size_t)i * M_ * D_;
            float* fc2_x_i  = fc2_x  + (size_t)i * M_ * DFF_;

            // mlp.pc_layer2 (linear fc2)
            layernorm_kernel<<<M_, 128, 0, stream>>>(nt, ln2g + i * D_, ln2b + i * D_, tmpD);
            gemm(fc2_x_i, W2_i, b2_i, tmpD2, M_, D_, DFF_, true, 1.f, 0.f);
            sub_kernel<<<gMD, 256, 0, stream>>>(tmpD, tmpD2, tmpD2, M_ * D_);  // e2
            gemm(tmpD2, W2_i, nullptr, fc2_x_i, M_, DFF_, D_, false, GAMMA_, 1.f);

            // mlp.pc_layer1 (fc1 + exact GELU), vjp
            gemm(fc1_x_i, W1_i, b1_i, tmpF, M_, DFF_, D_, true, 1.f, 0.f);     // z1
            gelu_vjp_kernel<<<gMF, 256, 0, stream>>>(tmpF, fc2_x_i, M_ * DFF_); // g1
            gemm(tmpF, W1_i, nullptr, fc1_x_i, M_, D_, DFF_, false, GAMMA_, 1.f);

            // attn.pc_output (output projection)
            layernorm_kernel<<<M_, 128, 0, stream>>>(fc1_x_i, ln1g + i * D_, ln1b + i * D_, tmpD);
            gemm(ao_x_i, Wo_i, bo_i, tmpD2, M_, D_, D_, true, 1.f, 0.f);
            sub_kernel<<<gMD, 256, 0, stream>>>(tmpD, tmpD2, tmpD2, M_ * D_);  // eo
            gemm(tmpD2, Wo_i, nullptr, ao_x_i, M_, D_, D_, false, GAMMA_, 1.f);

            // attn.pc_qkv: forward attention then J^T * ea
            gemm(attn_x_i, Wq_i, bq_i, qb, M_, D_, D_, true, 1.f, 0.f);
            gemm(attn_x_i, Wk_i, bk_i, kb, M_, D_, D_, true, 1.f, 0.f);
            gemm(attn_x_i, Wv_i, bv_i, vb, M_, D_, D_, true, 1.f, 0.f);
            attn_softmax_kernel<<<B_ * H_ * S_, 128, 0, stream>>>(qb, kb, pbuf);
            attn_pv_kernel<<<gMD, 256, 0, stream>>>(pbuf, vb, tmpD);           // mua
            sub_kernel<<<gMD, 256, 0, stream>>>(ao_x_i, tmpD, tmpD2, M_ * D_); // ea
            attn_dv_kernel<<<gMD, 256, 0, stream>>>(pbuf, tmpD2, dvb);
            attn_dp_kernel<<<gP, 256, 0, stream>>>(tmpD2, vb, dsb);
            attn_dsoftmax_kernel<<<B_ * H_ * S_, 128, 0, stream>>>(pbuf, dsb);
            attn_dq_kernel<<<gMD, 256, 0, stream>>>(dsb, kb, dqb);
            attn_dk_kernel<<<gMD, 256, 0, stream>>>(dsb, qb, dkb);
            gemm(dqb, Wq_i, nullptr, attn_x_i, M_, D_, D_, false, GAMMA_, 1.f);
            gemm(dkb, Wk_i, nullptr, attn_x_i, M_, D_, D_, false, GAMMA_, 1.f);
            gemm(dvb, Wv_i, nullptr, attn_x_i, M_, D_, D_, false, GAMMA_, 1.f);
            nt = attn_x_i;
        }
        // emb_x update is dead code for the returned logits -> skipped
    }

    // final logits -> d_out
    gemm(out_x, Wout, bout, (float*)d_out, M_, V_, D_, true, 1.f, 0.f);
}
